// CrossAttentionModule_73289321939572
// MI455X (gfx1250) — compile-verified, matrix operand order checked by reading, not run on hardware
//
#include <hip/hip_runtime.h>
#include <hip/hip_bf16.h>
#include <math.h>

// ---------------------------------------------------------------------------
// Problem constants (from the reference)
// ---------------------------------------------------------------------------
#define B_   2
#define NP_  384
#define NC_  96
#define ND_  128
#define PD_  128
#define DD_  32
#define IH_  32
#define H_   4
#define CH_  32
#define NPC_ (B_ * NP_ * NC_)          // 73728 pair rows
#define PAIRF ((size_t)NPC_ * 128)     // floats in one pair-sized tensor

typedef __attribute__((ext_vector_type(16))) _Float16 v16h;
typedef __attribute__((ext_vector_type(8)))  _Float16 v8h;
typedef __attribute__((ext_vector_type(4)))  _Float16 v4h;
typedef __attribute__((ext_vector_type(8)))  float    v8f;

// Combine the two contiguous 8-half runs of a CDNA5 f16 fragment.
__device__ __forceinline__ v16h frag16(v8h lo, v8h hi) {
    return __builtin_shufflevector(lo, hi, 0,1,2,3,4,5,6,7,8,9,10,11,12,13,14,15);
}

// K-index of element i of an f16 WMMA A/B fragment for lane half `hi`
__device__ __forceinline__ int kmap(int i, int hi) {
    return (i < 8) ? (hi * 8 + i) : (16 + hi * 8 + (i - 8));
}

__device__ __forceinline__ v4h cvt4(float4 v) {
    v4h h = { (_Float16)v.x, (_Float16)v.y, (_Float16)v.z, (_Float16)v.w };
    return h;
}

// ---------------------------------------------------------------------------
// WMMA GEMM:  Y[M,N] = act( scale * X[M,K] @ W[N,K]^T + bias[N] )
// Block = 128 threads (4 waves): tile 64(M) x (NBT*16)(N), K-chunks of 32
// staged in LDS as f16.  NBT is a template parameter so the accumulator
// array is fully register-resident (no scratch spills).
// act: 0 = none, 1 = sigmoid, 2 = relu
// ---------------------------------------------------------------------------
template<int NBT>
__global__ void gemm_wmma_kernel(const float* __restrict__ X,
                                 const float* __restrict__ W,
                                 const float* __restrict__ bias,
                                 float* __restrict__ Y,
                                 int M, int N, int K, int ldW,
                                 float scale, int act)
{
    constexpr int NB = NBT * 16;
    __shared__ _Float16 As[64][32];
    __shared__ _Float16 Bs[NB][32];

    const int t    = threadIdx.x;          // 0..127
    const int lane = t & 31;
    const int wid  = t >> 5;               // 0..3
    const int l15  = lane & 15;
    const int hi   = lane >> 4;
    const int nblk = blockIdx.x * NB;
    const int m0   = blockIdx.y * 64;
    const int mrow = wid * 16 + l15;

    v8f acc[NBT];
#pragma unroll
    for (int i = 0; i < NBT; ++i) acc[i] = (v8f){};

    for (int k0 = 0; k0 < K; k0 += 32) {
        __syncthreads();
        // stage A (64x32) and B (NBx32) tiles as f16, coalesced float4 loads
        for (int f = t; f < 64 * 8; f += 128) {
            const int row = f >> 3, c4 = (f & 7) * 4;
            *(v4h*)&As[row][c4] =
                cvt4(*(const float4*)(X + (size_t)(m0 + row) * K + k0 + c4));
        }
#pragma unroll
        for (int f = t; f < NB * 8; f += 128) {
            const int row = f >> 3, c4 = (f & 7) * 4;
            *(v4h*)&Bs[row][c4] =
                cvt4(*(const float4*)(W + (size_t)(nblk + row) * ldW + k0 + c4));
        }
        if (k0 + 32 < K)   // prefetch next X chunk (global_prefetch_b8)
            __builtin_prefetch(X + (size_t)(m0 + (t >> 1)) * K + k0 + 32, 0, 1);
        __syncthreads();

        const v16h a = frag16(*(const v8h*)&As[mrow][hi * 8],
                              *(const v8h*)&As[mrow][16 + hi * 8]);
#pragma unroll
        for (int nt = 0; nt < NBT; ++nt) {
            const v16h b = frag16(*(const v8h*)&Bs[nt * 16 + l15][hi * 8],
                                  *(const v8h*)&Bs[nt * 16 + l15][16 + hi * 8]);
            acc[nt] = __builtin_amdgcn_wmma_f32_16x16x32_f16(false, a, false, b,
                                                             (short)0, acc[nt],
                                                             false, false);
        }
    }

#pragma unroll
    for (int nt = 0; nt < NBT; ++nt) {
#pragma unroll
        for (int r = 0; r < 8; ++r) {
            const int row = m0 + wid * 16 + r + 8 * hi;
            const int col = nblk + nt * 16 + l15;
            float v = acc[nt][r] * scale;
            if (bias) v += bias[col];
            if (act == 1)      v = 1.0f / (1.0f + expf(-v));
            else if (act == 2) v = fmaxf(v, 0.0f);
            Y[(size_t)row * N + col] = v;
        }
    }
}

// ---------------------------------------------------------------------------
// LayerNorm over last dim == 128.  One wave per row (4 floats/lane).
// ---------------------------------------------------------------------------
__global__ void ln128_kernel(const float* __restrict__ x,
                             const float* __restrict__ g,
                             const float* __restrict__ b,
                             float* __restrict__ y, int rows)
{
    const int wid  = (blockIdx.x * blockDim.x + threadIdx.x) >> 5;
    const int lane = threadIdx.x & 31;
    if (wid >= rows) return;

    const float4 v  = ((const float4*)(x + (size_t)wid * 128))[lane];
    float s  = v.x + v.y + v.z + v.w;
    float s2 = v.x * v.x + v.y * v.y + v.z * v.z + v.w * v.w;
#pragma unroll
    for (int off = 1; off < 32; off <<= 1) {
        s  += __shfl_xor(s,  off);
        s2 += __shfl_xor(s2, off);
    }
    const float mean = s * (1.0f / 128.0f);
    const float var  = s2 * (1.0f / 128.0f) - mean * mean;
    const float rstd = rsqrtf(var + 1e-5f);

    const float4 gg = ((const float4*)g)[lane];
    const float4 bb = ((const float4*)b)[lane];
    float4 o;
    o.x = (v.x - mean) * rstd * gg.x + bb.x;
    o.y = (v.y - mean) * rstd * gg.y + bb.y;
    o.z = (v.z - mean) * rstd * gg.z + bb.z;
    o.w = (v.w - mean) * rstd * gg.w + bb.w;
    ((float4*)(y + (size_t)wid * 128))[lane] = o;
}

// ---------------------------------------------------------------------------
// Gated H=4 projection: out[row,h] = (x·w_h + b_h) * sigmoid(x·wg_h + bg_h)
// Weights staged in LDS once per block.
// ---------------------------------------------------------------------------
__global__ void biasgate_kernel(const float* __restrict__ X,
                                const float* __restrict__ w,
                                const float* __restrict__ bvec,
                                const float* __restrict__ wg,
                                const float* __restrict__ bgvec,
                                float* __restrict__ out, int rows, int K)
{
    __shared__ float wS[H_ * 128];
    __shared__ float wgS[H_ * 128];
    const int t = threadIdx.x;
    for (int i = t; i < H_ * K; i += blockDim.x) { wS[i] = w[i]; wgS[i] = wg[i]; }
    __syncthreads();

    const int row = blockIdx.x * blockDim.x + t;
    if (row >= rows) return;
    const float* xr = X + (size_t)row * K;
#pragma unroll
    for (int h = 0; h < H_; ++h) {
        float s  = bvec  ? bvec[h]  : 0.0f;
        float sg = bgvec ? bgvec[h] : 0.0f;
        for (int k = 0; k < K; ++k) {
            const float xv = xr[k];
            s  += xv * wS [h * K + k];
            sg += xv * wgS[h * K + k];
        }
        out[(size_t)row * H_ + h] = s * (1.0f / (1.0f + expf(-sg)));
    }
}

// ---------------------------------------------------------------------------
// Flash attention.  Block = NW waves sharing one (outer, head); wave w owns
// query tile blockIdx.x*NW + w.  Each 32-key chunk of K/V is staged once in
// LDS (V transposed) so all waves fragment it with ds_load_b128.
//   mask  idx = b*mb + r*mr + q*mq + k*mk   (b = o/div, r = o%div)
//   bias2 idx = b*sb + r*sr + h*sh + q*sq + k*sk
// ---------------------------------------------------------------------------
__global__ void flash_attn_kernel(const float* __restrict__ Qm,
                                  const float* __restrict__ Km,
                                  const float* __restrict__ Vm,
                                  float* __restrict__ Om,
                                  int Lq, int Lk, int odiv,
                                  const unsigned char* __restrict__ mask,
                                  int mb, int mr, int mq, int mk,
                                  const float* __restrict__ bias2,
                                  int sb, int sr, int sh, int sq, int sk)
{
    __shared__ _Float16 Ks[32][32];          // 2 KB
    __shared__ _Float16 VsT[32][32];         // 2 KB  (V transposed: [ch][key])
    __shared__ _Float16 Plds[8][16][32];     // 8 KB  (per-wave prob staging)

    const int t    = threadIdx.x;
    const int NT   = blockDim.x;
    const int lane = t & 31;
    const int wid  = t >> 5;
    const int l15  = lane & 15;
    const int hi   = lane >> 4;
    const int q0   = (blockIdx.x * (NT >> 5) + wid) * 16;
    const int h    = blockIdx.y;
    const int o    = blockIdx.z;
    const int bidx = o / odiv;
    const int rr   = o % odiv;

    const float* Qb = Qm + ((size_t)o * Lq) * 128 + h * CH_;
    const float* Kb = Km + ((size_t)o * Lk) * 128 + h * CH_;
    const float* Vb = Vm + ((size_t)o * Lk) * 128 + h * CH_;

    // Q fragment (A operand), loaded once per wave
    v16h aq;
#pragma unroll
    for (int i = 0; i < 16; ++i)
        aq[i] = (_Float16)Qb[(size_t)(q0 + l15) * 128 + kmap(i, hi)];

    float rmax[8], rsum[8];
    v8f oacc0 = {}, oacc1 = {};
#pragma unroll
    for (int r = 0; r < 8; ++r) { rmax[r] = -1e30f; rsum[r] = 0.0f; }

    for (int kc = 0; kc < Lk; kc += 32) {
        __syncthreads();
        // stage K (row-major) and V (transposed) chunk as f16
        for (int f = t; f < 32 * 8; f += NT) {
            const int key = f >> 3, c4 = (f & 7) * 4;
            *(v4h*)&Ks[key][c4] =
                cvt4(*(const float4*)(Kb + (size_t)(kc + key) * 128 + c4));
            const float4 vv = *(const float4*)(Vb + (size_t)(kc + key) * 128 + c4);
            VsT[c4 + 0][key] = (_Float16)vv.x;
            VsT[c4 + 1][key] = (_Float16)vv.y;
            VsT[c4 + 2][key] = (_Float16)vv.z;
            VsT[c4 + 3][key] = (_Float16)vv.w;
        }
        if (kc + 32 < Lk) {   // prefetch next K/V chunk
            __builtin_prefetch(Kb + (size_t)(kc + 32 + (t >> 3)) * 128, 0, 1);
            __builtin_prefetch(Vb + (size_t)(kc + 32 + (t >> 3)) * 128, 0, 1);
        }
        __syncthreads();

        // two 16x16 score tiles for this 32-key chunk
        v8f s[2];
#pragma unroll
        for (int tt = 0; tt < 2; ++tt) {
            const v16h bk = frag16(*(const v8h*)&Ks[tt * 16 + l15][hi * 8],
                                   *(const v8h*)&Ks[tt * 16 + l15][16 + hi * 8]);
            v8f z = {};
            s[tt] = __builtin_amdgcn_wmma_f32_16x16x32_f16(false, aq, false, bk,
                                                           (short)0, z, false, false);
            const int key = kc + tt * 16 + l15;
#pragma unroll
            for (int r = 0; r < 8; ++r) {
                const int qi = q0 + r + 8 * hi;
                float bv = 0.0f;
                if (mask)
                    bv += mask[(size_t)bidx * mb + (size_t)rr * mr +
                               (size_t)qi * mq + (size_t)key * mk] ? 0.0f : -1e9f;
                if (bias2)
                    bv += bias2[(size_t)bidx * sb + (size_t)rr * sr + (size_t)h * sh +
                                (size_t)qi * sq + (size_t)key * sk];
                s[tt][r] += bv;
            }
        }

        // online softmax (row lives in the 16 lanes of the half-group)
#pragma unroll
        for (int r = 0; r < 8; ++r) {
            float cm = fmaxf(s[0][r], s[1][r]);
#pragma unroll
            for (int off = 1; off < 16; off <<= 1) cm = fmaxf(cm, __shfl_xor(cm, off));
            const float nm    = fmaxf(rmax[r], cm);
            const float alpha = expf(rmax[r] - nm);
            rmax[r] = nm;
            const float p0 = expf(s[0][r] - nm);
            const float p1 = expf(s[1][r] - nm);
            s[0][r] = p0; s[1][r] = p1;
            float cs = p0 + p1;
#pragma unroll
            for (int off = 1; off < 16; off <<= 1) cs += __shfl_xor(cs, off);
            rsum[r]  = rsum[r] * alpha + cs;
            oacc0[r] *= alpha;
            oacc1[r] *= alpha;
        }

        // stage probabilities (wave-local slice), re-fragment as A operand
#pragma unroll
        for (int r = 0; r < 8; ++r) {
            Plds[wid][r + 8 * hi][l15]      = (_Float16)s[0][r];
            Plds[wid][r + 8 * hi][16 + l15] = (_Float16)s[1][r];
        }
        const v16h ap = frag16(*(const v8h*)&Plds[wid][l15][hi * 8],
                               *(const v8h*)&Plds[wid][l15][16 + hi * 8]);

        // V fragments for the two 16-channel halves (from transposed LDS)
        const v16h bv0 = frag16(*(const v8h*)&VsT[l15][hi * 8],
                                *(const v8h*)&VsT[l15][16 + hi * 8]);
        const v16h bv1 = frag16(*(const v8h*)&VsT[16 + l15][hi * 8],
                                *(const v8h*)&VsT[16 + l15][16 + hi * 8]);
        oacc0 = __builtin_amdgcn_wmma_f32_16x16x32_f16(false, ap, false, bv0,
                                                       (short)0, oacc0, false, false);
        oacc1 = __builtin_amdgcn_wmma_f32_16x16x32_f16(false, ap, false, bv1,
                                                       (short)0, oacc1, false, false);
    }

#pragma unroll
    for (int r = 0; r < 8; ++r) {
        const float inv = 1.0f / rsum[r];
        const int   qi  = q0 + r + 8 * hi;
        float* op = Om + ((size_t)o * Lq + qi) * 128 + h * CH_;
        op[l15]      = oacc0[r] * inv;
        op[16 + l15] = oacc1[r] * inv;
    }
}

// ---------------------------------------------------------------------------
// Elementwise helpers
// ---------------------------------------------------------------------------
__global__ void mul_kernel(float* __restrict__ x, const float* __restrict__ y, int n)
{
    for (int i = blockIdx.x * blockDim.x + threadIdx.x; i < n; i += gridDim.x * blockDim.x)
        x[i] *= y[i];
}

__global__ void add_kernel(float* __restrict__ x, const float* __restrict__ y, int n)
{
    for (int i = blockIdx.x * blockDim.x + threadIdx.x; i < n; i += gridDim.x * blockDim.x)
        x[i] += y[i];
}

// out[outRow] = (z ? z[row] : 0) + y[row] * mask   (128 floats per row)
__global__ void resid_mask_kernel(const float* __restrict__ z,
                                  const float* __restrict__ y,
                                  const unsigned char* __restrict__ mask,
                                  float* __restrict__ out,
                                  int n1, int n2,
                                  int mb, int mx, int my,
                                  int ob, int ox, int oy)
{
    const int row = blockIdx.x;
    const int c   = threadIdx.x;
    const int b   = row / (n1 * n2);
    const int x   = (row / n2) % n1;
    const int yj  = row % n2;
    float m = 1.0f;
    if (mask)
        m = mask[(size_t)b * mb + (size_t)x * mx + (size_t)yj * my] ? 1.0f : 0.0f;
    float v = y[(size_t)row * 128 + c] * m;
    if (z) v += z[(size_t)row * 128 + c];
    const size_t orow = (size_t)b * ob + (size_t)x * ox + (size_t)yj * oy;
    out[orow * 128 + c] = v;
}

// pair[b,i,j,:] = pairIn[b,i,j,:] + pp[b,i,:] + cc[b,j,:] + bo[:]
__global__ void add_inter_kernel(const float* __restrict__ pin,
                                 const float* __restrict__ pp,
                                 const float* __restrict__ cc,
                                 const float* __restrict__ bo,
                                 float* __restrict__ out)
{
    const int row = blockIdx.x;
    const int c   = threadIdx.x;
    const int b   = row / (NP_ * NC_);
    const int i   = (row / NC_) % NP_;
    const int j   = row % NC_;
    out[(size_t)row * 128 + c] = pin[(size_t)row * 128 + c]
                               + pp[((size_t)b * NP_ + i) * 128 + c]
                               + cc[((size_t)b * NC_ + j) * 128 + c]
                               + bo[c];
}

// ---------------------------------------------------------------------------
// Host orchestration
// ---------------------------------------------------------------------------
static inline void gemm(hipStream_t s, const float* X, const float* W, const float* bias,
                        float* Y, int M, int N, int K, int ldW, float scale, int act)
{
    if (N == 32) {
        dim3 g(1, M / 64);
        gemm_wmma_kernel<2><<<g, 128, 0, s>>>(X, W, bias, Y, M, N, K, ldW, scale, act);
    } else {
        dim3 g(N / 128, M / 64);
        gemm_wmma_kernel<8><<<g, 128, 0, s>>>(X, W, bias, Y, M, N, K, ldW, scale, act);
    }
}

static inline void ln(hipStream_t s, const float* x, const float* g, const float* b,
                      float* y, int rows)
{
    ln128_kernel<<<(rows * 32 + 255) / 256, 256, 0, s>>>(x, g, b, y, rows);
}

extern "C" void kernel_launch(void* const* d_in, const int* in_sizes, int n_in,
                              void* d_out, int out_size, void* d_ws, size_t ws_size,
                              hipStream_t stream)
{
    (void)in_sizes; (void)n_in; (void)out_size; (void)ws_size;
    const float QSC = 0.17677669529663687f;   // 1/sqrt(CH)

    // ---- inputs (setup_inputs dict order) -------------------------------
    const float*         p_embed  = (const float*)d_in[0];
    const unsigned char* p_mask   = (const unsigned char*)d_in[1];
    const float*         c_embed  = (const float*)d_in[2];
    const unsigned char* c_mask   = (const unsigned char*)d_in[3];
    const float*         pair_in  = (const float*)d_in[4];
    const unsigned char* pm       = (const unsigned char*)d_in[5];
    const float*         ccd      = (const float*)d_in[6];
    const float*         ppd      = (const float*)d_in[7];

    // ---- params, JAX tree order (dict keys sorted alphabetically) -------
    const float* const* P = (const float* const*)(d_in + 8);
    const int CR  = 0;   // c_row: lni_b,lni_g,lnj_b,lnj_g,lnp_b,lnp_g,
                         //        mha.bg,mha.bo,mha.wg,mha.wk,mha.wo,mha.wq,mha.wv, w, wg
    const int CT  = 15;  // c_tr: b1,b2,ln_b,ln_g,w1,w2
    const int IN  = 21;  // inter: bc,bo,bp,lnc_b,lnc_g,lnp_b,lnp_g,wc,wo,wp
    const int PR  = 31;  // p_row (same layout as c_row)
    const int PT_ = 46;  // p_tr
    const int QT  = 52;  // pair_tr
    const int TR  = 58;  // tri: b,bg,ln_b,ln_g,mha.bg,mha.bo,mha.wg,mha.wk,
                         //      mha.wo,mha.wq,mha.wv, w, wg

    // ---- workspace layout (floats) --------------------------------------
    float* ws = (float*)d_ws;
    float* Z  = ws + 0 * PAIRF;
    float* Q  = ws + 1 * PAIRF;       // Q..Kb contiguous -> doubles as NPCx256
    float* Kb = ws + 2 * PAIRF;
    float* V  = ws + 3 * PAIRF;
    float* G  = ws + 4 * PAIRF;
    float* O  = ws + 5 * PAIRF;
    float* Y  = ws + 6 * PAIRF;
    float* PA = ws + 7 * PAIRF;       // pair after tri1 / tri2
    float* PTb= ws + 8 * PAIRF;       // transposed pair, later pair+inter
    float* sm = ws + 9 * PAIRF;
    float* DB1 = sm;                    sm += B_ * NC_ * NC_ * H_;
    float* DB2 = sm;                    sm += B_ * NP_ * NP_ * H_;
    float* PBP = sm;                    sm += (size_t)NPC_ * H_;
    float* PBC = sm;                    sm += (size_t)NPC_ * H_;
    float* NI  = sm;                    sm += B_ * NP_ * 128;
    float* NJ  = sm;                    sm += B_ * NC_ * 128;
    float* CI  = sm;                    sm += B_ * NC_ * 128;
    float* CJ  = sm;                    sm += B_ * NP_ * 128;
    float* P32 = sm;                    sm += B_ * NP_ * IH_;
    float* C32 = sm;                    sm += B_ * NC_ * IH_;
    float* PPb = sm;                    sm += B_ * NP_ * 128;
    float* CCb = sm;                    sm += B_ * NC_ * 128;

    // outputs: p_e | c_e | pair (concatenated)
    float* PE   = (float*)d_out;
    float* CE   = PE + B_ * NP_ * 128;
    float* Pout = CE + B_ * NC_ * 128;

    const int PAIRN = (int)PAIRF;

    // ================= TRI pass 1 (over j = NC) ==========================
    ln(stream, pair_in, P[TR+3], P[TR+2], Z, NPC_);
    gemm(stream, Z, P[TR+9], nullptr, Q,  NPC_, 128, 128, 128, QSC,  0);  // wq
    gemm(stream, Z, P[TR+7], nullptr, Kb, NPC_, 128, 128, 128, 1.0f, 0);  // wk
    gemm(stream, Z, P[TR+10],nullptr, V,  NPC_, 128, 128, 128, 1.0f, 0);  // wv
    gemm(stream, Z, P[TR+6], P[TR+4], G,  NPC_, 128, 128, 128, 1.0f, 1);  // wg+bg
    biasgate_kernel<<<(B_*NC_*NC_ + 255)/256, 256, 0, stream>>>(
        ccd, P[TR+11], P[TR+0], P[TR+12], P[TR+1], DB1, B_*NC_*NC_, DD_);
    flash_attn_kernel<<<dim3(1, H_, B_*NP_), 6*32, 0, stream>>>(
        Q, Kb, V, O, NC_, NC_, NP_,
        pm,  NP_*NC_, NC_, 0, 1,
        DB1, NC_*NC_*H_, 0, 1, H_, NC_*H_);
    mul_kernel<<<4096, 256, 0, stream>>>(O, G, PAIRN);
    gemm(stream, O, P[TR+8], P[TR+5], Y, NPC_, 128, 128, 128, 1.0f, 0);   // wo+bo
    resid_mask_kernel<<<NPC_, 128, 0, stream>>>(Z, Y, pm, PA,
        NP_, NC_, NP_*NC_, NC_, 1, NP_*NC_, NC_, 1);

    // ================= TRI pass 2 (transposed, over i = NP) ==============
    resid_mask_kernel<<<NPC_, 128, 0, stream>>>(nullptr, PA, nullptr, PTb,
        NP_, NC_, 0, 0, 0, NP_*NC_, 1, NP_);                   // transpose
    ln(stream, PTb, P[TR+3], P[TR+2], Z, NPC_);
    gemm(stream, Z, P[TR+9], nullptr, Q,  NPC_, 128, 128, 128, QSC,  0);
    gemm(stream, Z, P[TR+7], nullptr, Kb, NPC_, 128, 128, 128, 1.0f, 0);
    gemm(stream, Z, P[TR+10],nullptr, V,  NPC_, 128, 128, 128, 1.0f, 0);
    gemm(stream, Z, P[TR+6], P[TR+4], G,  NPC_, 128, 128, 128, 1.0f, 1);
    biasgate_kernel<<<(B_*NP_*NP_ + 255)/256, 256, 0, stream>>>(
        ppd, P[TR+11], P[TR+0], P[TR+12], P[TR+1], DB2, B_*NP_*NP_, DD_);
    flash_attn_kernel<<<dim3(3, H_, B_*NC_), 8*32, 0, stream>>>(
        Q, Kb, V, O, NP_, NP_, NC_,
        pm,  NP_*NC_, 1, 0, NC_,
        DB2, NP_*NP_*H_, 0, 1, H_, NP_*H_);
    mul_kernel<<<4096, 256, 0, stream>>>(O, G, PAIRN);
    gemm(stream, O, P[TR+8], P[TR+5], Y, NPC_, 128, 128, 128, 1.0f, 0);
    resid_mask_kernel<<<NPC_, 128, 0, stream>>>(Z, Y, pm, PA,
        NC_, NP_, NP_*NC_, 1, NC_, NP_*NC_, 1, NC_);

    // ================= p_row attention ===================================
    ln(stream, p_embed, P[PR+1], P[PR+0], NI, B_*NP_);
    ln(stream, c_embed, P[PR+3], P[PR+2], NJ, B_*NC_);
    ln(stream, PA, P[PR+5], P[PR+4], Z, NPC_);
    biasgate_kernel<<<(NPC_ + 255)/256, 256, 0, stream>>>(
        Z, P[PR+13], nullptr, P[PR+14], nullptr, PBP, NPC_, 128);
    gemm(stream, NI, P[PR+11], nullptr, Q,  B_*NP_, 128, 128, 128, QSC,  0);
    gemm(stream, NJ, P[PR+9],  nullptr, Kb, B_*NC_, 128, 128, 128, 1.0f, 0);
    gemm(stream, NJ, P[PR+12], nullptr, V,  B_*NC_, 128, 128, 128, 1.0f, 0);
    gemm(stream, NI, P[PR+8],  P[PR+6], G,  B_*NP_, 128, 128, 128, 1.0f, 1);
    flash_attn_kernel<<<dim3(3, H_, B_), 8*32, 0, stream>>>(
        Q, Kb, V, O, NP_, NC_, 1,
        pm,  NP_*NC_, 0, NC_, 1,
        PBP, NP_*NC_*H_, 0, 1, NC_*H_, H_);
    mul_kernel<<<256, 256, 0, stream>>>(O, G, B_*NP_*128);
    gemm(stream, O, P[PR+10], P[PR+7], Y, B_*NP_, 128, 128, 128, 1.0f, 0);
    resid_mask_kernel<<<B_*NP_, 128, 0, stream>>>(NI, Y, p_mask, PE,
        NP_, 1, NP_, 1, 0, NP_, 1, 0);

    // ================= c_row attention (kv = p_e, pair transposed) =======
    ln(stream, c_embed, P[CR+1], P[CR+0], CI, B_*NC_);
    ln(stream, PE,      P[CR+3], P[CR+2], CJ, B_*NP_);
    ln(stream, PA, P[CR+5], P[CR+4], Z, NPC_);
    biasgate_kernel<<<(NPC_ + 255)/256, 256, 0, stream>>>(
        Z, P[CR+13], nullptr, P[CR+14], nullptr, PBC, NPC_, 128);
    gemm(stream, CI, P[CR+11], nullptr, Q,  B_*NC_, 128, 128, 128, QSC,  0);
    gemm(stream, CJ, P[CR+9],  nullptr, Kb, B_*NP_, 128, 128, 128, 1.0f, 0);
    gemm(stream, CJ, P[CR+12], nullptr, V,  B_*NP_, 128, 128, 128, 1.0f, 0);
    gemm(stream, CI, P[CR+8],  P[CR+6], G,  B_*NC_, 128, 128, 128, 1.0f, 1);
    flash_attn_kernel<<<dim3(1, H_, B_), 6*32, 0, stream>>>(
        Q, Kb, V, O, NC_, NP_, 1,
        pm,  NP_*NC_, 0, 1, NC_,
        PBC, NP_*NC_*H_, 0, 1, H_, NC_*H_);
    mul_kernel<<<64, 256, 0, stream>>>(O, G, B_*NC_*128);
    gemm(stream, O, P[CR+10], P[CR+7], Y, B_*NC_, 128, 128, 128, 1.0f, 0);
    resid_mask_kernel<<<B_*NC_, 128, 0, stream>>>(CI, Y, c_mask, CE,
        NC_, 1, NC_, 1, 0, NC_, 1, 0);

    // ================= transitions on p_e / c_e ==========================
    ln(stream, PE, P[PT_+3], P[PT_+2], NI, B_*NP_);
    gemm(stream, NI, P[PT_+4], P[PT_+0], Q, B_*NP_, 256, 128, 128, 1.0f, 2);
    gemm(stream, Q,  P[PT_+5], P[PT_+1], Y, B_*NP_, 128, 256, 256, 1.0f, 0);
    add_kernel<<<384, 256, 0, stream>>>(PE, Y, B_*NP_*128);

    ln(stream, CE, P[CT+3], P[CT+2], NJ, B_*NC_);
    gemm(stream, NJ, P[CT+4], P[CT+0], Q, B_*NC_, 256, 128, 128, 1.0f, 2);
    gemm(stream, Q,  P[CT+5], P[CT+1], Y, B_*NC_, 128, 256, 256, 1.0f, 0);
    add_kernel<<<96, 256, 0, stream>>>(CE, Y, B_*NC_*128);

    // ================= inter (outer-product pair update) =================
    ln(stream, PE, P[IN+6], P[IN+5], NI, B_*NP_);
    ln(stream, CE, P[IN+4], P[IN+3], NJ, B_*NC_);
    gemm(stream, NI, P[IN+9], P[IN+2], P32, B_*NP_, IH_, 128, 128, 1.0f, 0);
    gemm(stream, NJ, P[IN+7], P[IN+0], C32, B_*NC_, IH_, 128, 128, 1.0f, 0);
    gemm(stream, P32, P[IN+8],       nullptr, PPb, B_*NP_, 128, IH_, 2*IH_, 1.0f, 0);
    gemm(stream, C32, P[IN+8] + IH_, nullptr, CCb, B_*NC_, 128, IH_, 2*IH_, 1.0f, 0);
    add_inter_kernel<<<NPC_, 128, 0, stream>>>(PA, PPb, CCb, P[IN+1], PTb);

    // ================= final pair transition (no residual) ===============
    ln(stream, PTb, P[QT+3], P[QT+2], Z, NPC_);
    gemm(stream, Z, P[QT+4], P[QT+0], Q, NPC_, 256, 128, 128, 1.0f, 2);
    gemm(stream, Q, P[QT+5], P[QT+1], Y, NPC_, 128, 256, 256, 1.0f, 0);
    resid_mask_kernel<<<NPC_, 128, 0, stream>>>(nullptr, Y, pm, Pout,
        NP_, NC_, NP_*NC_, NC_, 1, NP_*NC_, NC_, 1);
}
